// SelfAttention_2619930050612
// MI455X (gfx1250) — matrix operand order repaired and checked by reading, not compiled, not run on hardware
//
#include <hip/hip_runtime.h>
#include <hip/hip_bf16.h>

#define EMB 512
#define SEQ 4096
#define BATCH 2
#define NHEADS 8
#define HD 64
#define ROWS (BATCH*SEQ)   // 8192

typedef __attribute__((ext_vector_type(16))) __bf16 v16bf;
typedef __attribute__((ext_vector_type(8)))  float  v8f;

union ABfrag { uint4 u[2]; v16bf v; };

__device__ __forceinline__ unsigned short f2bfu(float f) {
    unsigned u = __builtin_bit_cast(unsigned, f);
    u += 0x7FFFu + ((u >> 16) & 1u);            // round-to-nearest-even
    return (unsigned short)(u >> 16);
}

// async copy of 16 bytes: global -> LDS, tracked by ASYNCcnt.
// IOFFSET applies to BOTH the LDS dest VGPR and the global addr (ISA 10.2/15.18).
#define GLDS128(ldsa, ga, off)                                              \
    asm volatile("global_load_async_to_lds_b128 %0, %1, off offset:" #off   \
                 :: "v"(ldsa), "v"(ga) : "memory")

__device__ __forceinline__ void wait_async_le16() {
#if __has_builtin(__builtin_amdgcn_s_wait_asynccnt)
    __builtin_amdgcn_s_wait_asynccnt(16);
#else
    asm volatile("s_wait_asynccnt 0x10" ::: "memory");
#endif
}
__device__ __forceinline__ void wait_async_0() {
#if __has_builtin(__builtin_amdgcn_s_wait_asynccnt)
    __builtin_amdgcn_s_wait_asynccnt(0);
#else
    asm volatile("s_wait_asynccnt 0x0" ::: "memory");
#endif
}

// ---------------- conversion kernels ----------------
__global__ void cvt_bf16_kernel(const float* __restrict__ in,
                                unsigned short* __restrict__ out, int n) {
    for (int i = blockIdx.x * blockDim.x + threadIdx.x; i < n;
         i += gridDim.x * blockDim.x)
        out[i] = f2bfu(in[i]);
}

// W [K=512][N=512] f32  ->  Wt [N][K] bf16
__global__ void cvtT512_kernel(const float* __restrict__ W,
                               unsigned short* __restrict__ Wt) {
    const int n = blockIdx.x;
    for (int k = threadIdx.x; k < EMB; k += blockDim.x)
        Wt[n * EMB + k] = f2bfu(W[k * EMB + n]);
}

// ---------------- WMMA GEMM: 32x64 tile per wave ----------------
// mode 0: outb = bf16, head-major [b][h][n][64]   (Q, K)
// mode 1: outb = bf16, transposed [b][h][64][n]   (V^T)
// mode 2: outf = f32,  [m][512] = acc + bias + resid   (out-proj + residual)
__global__ void gemm512_kernel(const unsigned short* __restrict__ A,
                               const unsigned short* __restrict__ Wt,
                               const float* __restrict__ bias,
                               const float* __restrict__ resid,
                               unsigned short* __restrict__ outb,
                               float* __restrict__ outf,
                               int mode) {
    const int wave  = (int)((blockIdx.x * blockDim.x + threadIdx.x) >> 5);
    const int lane  = (int)(threadIdx.x & 31);
    const int mtile = wave >> 3;          // 256 row-tiles of 32
    const int ngrp  = wave & 7;           // 8 groups of 64 columns
    const int row0  = mtile * 32;
    const int ncol0 = ngrp * 64;
    const int half  = lane >> 4;
    const int l16   = lane & 15;

    v8f acc[2][4] = {};
    for (int k0 = 0; k0 < EMB; k0 += 32) {
        ABfrag a0, a1;
        const unsigned short* ar0 = A + (size_t)(row0 + l16) * EMB + k0;
        const unsigned short* ar1 = A + (size_t)(row0 + 16 + l16) * EMB + k0;
        a0.u[0] = *(const uint4*)(ar0 + half * 8);
        a0.u[1] = *(const uint4*)(ar0 + 16 + half * 8);
        a1.u[0] = *(const uint4*)(ar1 + half * 8);
        a1.u[1] = *(const uint4*)(ar1 + 16 + half * 8);
#pragma unroll
        for (int t = 0; t < 4; ++t) {
            ABfrag b;
            const unsigned short* brow =
                Wt + (size_t)(ncol0 + t * 16 + l16) * EMB + k0 + half * 8;
            b.u[0] = *(const uint4*)(brow);
            b.u[1] = *(const uint4*)(brow + 16);
            acc[0][t] = __builtin_amdgcn_wmma_f32_16x16x32_bf16(
                false, a0.v, false, b.v, (short)0, acc[0][t], false, false);
            acc[1][t] = __builtin_amdgcn_wmma_f32_16x16x32_bf16(
                false, a1.v, false, b.v, (short)0, acc[1][t], false, false);
        }
    }

#pragma unroll
    for (int mi = 0; mi < 2; ++mi) {
        const int rbase = row0 + mi * 16;
#pragma unroll
        for (int t = 0; t < 4; ++t) {
            const int col  = ncol0 + t * 16 + l16;
            const float bval = bias[col];
            if (mode == 0) {
                const int h = col >> 6, d = col & 63;
#pragma unroll
                for (int r = 0; r < 8; ++r) {
                    const int m = rbase + half * 8 + r;
                    const int b = m >> 12, n = m & 4095;
                    outb[((size_t)(b * NHEADS + h) * SEQ + n) * HD + d] =
                        f2bfu(acc[mi][t][r] + bval);
                }
            } else if (mode == 1) {
                const int h = col >> 6, d = col & 63;
                alignas(16) unsigned short tmp[8];
#pragma unroll
                for (int r = 0; r < 8; ++r) tmp[r] = f2bfu(acc[mi][t][r] + bval);
                const int m0 = rbase + half * 8;
                const int b = m0 >> 12, n0 = m0 & 4095;
                *(uint4*)(outb + ((size_t)(b * NHEADS + h) * HD + d) * SEQ + n0) =
                    *(const uint4*)tmp;
            } else {
#pragma unroll
                for (int r = 0; r < 8; ++r) {
                    const int m = rbase + half * 8 + r;
                    const size_t idx = (size_t)m * EMB + col;
                    outf[idx] = acc[mi][t][r] + bval + resid[idx];
                }
            }
        }
    }
}

// ---------------- flash attention ----------------
// One wave per (b,h, 16-row Q block). 2 waves/block.
// K,V tiles double-buffered in LDS via global_load_async_to_lds_b128.
// Per-wave LDS (ushorts): Kbuf[2][32][64]=4096, Vbuf[2][64][32]=4096, Pbuf[16][32]=512
#define WLDS 8704
__global__ void attn_kernel(const unsigned short* __restrict__ Q,
                            const unsigned short* __restrict__ K,
                            const unsigned short* __restrict__ Vt,
                            unsigned short* __restrict__ ctx) {
    __shared__ __align__(16) unsigned short sh[2 * WLDS];
    const int wib  = threadIdx.x >> 5;
    const int lane = threadIdx.x & 31;
    const int gw   = blockIdx.x * 2 + wib;
    const int bh   = gw >> 8;               // 0..15
    const int q0   = (gw & 255) * 16;
    const int half = lane >> 4, l16 = lane & 15;

    unsigned short* Kb = sh + wib * WLDS;           // [2][32][64]
    unsigned short* Vb = Kb + 4096;                 // [2][64][32]
    unsigned short* Pb = Kb + 8192;                 // [16][32]

    const unsigned short* Kg = K  + (size_t)bh * SEQ * HD;   // [n][64]
    const unsigned short* Vg = Vt + (size_t)bh * HD * SEQ;   // [d][4096]

    // Q fragments (A layout), resident for the whole pass
    ABfrag qf[2];
#pragma unroll
    for (int ds = 0; ds < 2; ++ds) {
        const unsigned short* ar = Q + ((size_t)bh * SEQ + q0 + l16) * HD + ds * 32;
        qf[ds].u[0] = *(const uint4*)(ar + half * 8);
        qf[ds].u[1] = *(const uint4*)(ar + 16 + half * 8);
    }

    // async stage of one 32-key tile into buffer bi (16 async b128 ops per lane)
    auto stage = [&](int k0, int bi) {
        // K tile: lane copies key-row k0+lane (128 B)
        unsigned long long gK = (unsigned long long)(Kg + (size_t)(k0 + lane) * HD);
        unsigned lK = (unsigned)(unsigned long long)(Kb + bi * 2048 + lane * 64);
        GLDS128(lK, gK, 0);  GLDS128(lK, gK, 16); GLDS128(lK, gK, 32); GLDS128(lK, gK, 48);
        GLDS128(lK, gK, 64); GLDS128(lK, gK, 80); GLDS128(lK, gK, 96); GLDS128(lK, gK, 112);
        // V tile: lane copies d-rows 2*lane and 2*lane+1 (64 B each)
        const int d0 = 2 * lane;
        unsigned long long gV0 = (unsigned long long)(Vg + (size_t)d0 * SEQ + k0);
        unsigned long long gV1 = (unsigned long long)(Vg + (size_t)(d0 + 1) * SEQ + k0);
        unsigned lV0 = (unsigned)(unsigned long long)(Vb + bi * 2048 + d0 * 32);
        unsigned lV1 = (unsigned)(unsigned long long)(Vb + bi * 2048 + (d0 + 1) * 32);
        GLDS128(lV0, gV0, 0); GLDS128(lV0, gV0, 16); GLDS128(lV0, gV0, 32); GLDS128(lV0, gV0, 48);
        GLDS128(lV1, gV1, 0); GLDS128(lV1, gV1, 16); GLDS128(lV1, gV1, 32); GLDS128(lV1, gV1, 48);
    };

    v8f cacc[4] = {};
    float marr[8], larr[8];
#pragma unroll
    for (int r = 0; r < 8; ++r) { marr[r] = -1e30f; larr[r] = 0.f; }

    stage(0, 0);

    for (int k0 = 0; k0 < SEQ; k0 += 32) {
        const int bi = (k0 >> 5) & 1;
        const bool more = (k0 + 32) < SEQ;
        if (more) { stage(k0 + 32, bi ^ 1); wait_async_le16(); }
        else      { wait_async_0(); }

        const unsigned short* kbb = Kb + bi * 2048;
        const unsigned short* vbb = Vb + bi * 2048;

        // S = Q K^T (raw, scaled later); two 16x16 tiles over 32 keys
        v8f s0 = {}, s1 = {};
#pragma unroll
        for (int ds = 0; ds < 2; ++ds) {
            ABfrag kf0, kf1;
            const unsigned short* b0 = kbb + (size_t)l16 * HD + ds * 32 + half * 8;
            const unsigned short* b1 = kbb + (size_t)(16 + l16) * HD + ds * 32 + half * 8;
            kf0.u[0] = *(const uint4*)(b0); kf0.u[1] = *(const uint4*)(b0 + 16);
            kf1.u[0] = *(const uint4*)(b1); kf1.u[1] = *(const uint4*)(b1 + 16);
            s0 = __builtin_amdgcn_wmma_f32_16x16x32_bf16(
                false, qf[ds].v, false, kf0.v, (short)0, s0, false, false);
            s1 = __builtin_amdgcn_wmma_f32_16x16x32_bf16(
                false, qf[ds].v, false, kf1.v, (short)0, s1, false, false);
        }

        // register-space online softmax: row M=r+half*8 lives on the 16 lanes of
        // this half, so reduce with xor masks 1..8 (stays inside the half).
#pragma unroll
        for (int r = 0; r < 8; ++r) {
            float mx = fmaxf(s0[r], s1[r]);
            mx = fmaxf(mx, __shfl_xor(mx, 1, 32));
            mx = fmaxf(mx, __shfl_xor(mx, 2, 32));
            mx = fmaxf(mx, __shfl_xor(mx, 4, 32));
            mx = fmaxf(mx, __shfl_xor(mx, 8, 32));
            mx *= 0.125f;                                   // 1/sqrt(64)
            const float mn = fmaxf(marr[r], mx);
            const float sc = __expf(marr[r] - mn);
            marr[r] = mn;
            const float p0 = __expf(s0[r] * 0.125f - mn);
            const float p1 = __expf(s1[r] * 0.125f - mn);
            float rs = p0 + p1;
            rs += __shfl_xor(rs, 1, 32);
            rs += __shfl_xor(rs, 2, 32);
            rs += __shfl_xor(rs, 4, 32);
            rs += __shfl_xor(rs, 8, 32);
            larr[r] = larr[r] * sc + rs;
#pragma unroll
            for (int t = 0; t < 4; ++t) cacc[t][r] *= sc;
            Pb[(r + half * 8) * 32 + l16]      = f2bfu(p0);
            Pb[(r + half * 8) * 32 + 16 + l16] = f2bfu(p1);
        }

        // P fragment (A layout) from LDS; DS ops are in-order within a wave
        ABfrag pf;
        pf.u[0] = *(const uint4*)(Pb + l16 * 32 + half * 8);
        pf.u[1] = *(const uint4*)(Pb + l16 * 32 + 16 + half * 8);

        // context += P x V (V^T rows give contiguous B fragments from LDS)
#pragma unroll
        for (int t = 0; t < 4; ++t) {
            ABfrag vf;
            const unsigned short* br = vbb + (size_t)(t * 16 + l16) * 32 + half * 8;
            vf.u[0] = *(const uint4*)(br);
            vf.u[1] = *(const uint4*)(br + 16);
            cacc[t] = __builtin_amdgcn_wmma_f32_16x16x32_bf16(
                false, pf.v, false, vf.v, (short)0, cacc[t], false, false);
        }
    }

    const int b = bh >> 3, h = bh & 7;
#pragma unroll
    for (int r = 0; r < 8; ++r) {
        const float inv = 1.f / larr[r];
        const int m = q0 + r + half * 8;
        const size_t base = ((size_t)b * SEQ + m) * EMB + h * HD;
#pragma unroll
        for (int t = 0; t < 4; ++t)
            ctx[base + t * 16 + l16] = f2bfu(cacc[t][r] * inv);
    }
}

// ---------------- LayerNorm: one 256-thread block per row ----------------
__global__ void ln_kernel(const float* __restrict__ t,
                          const float* __restrict__ gamma,
                          const float* __restrict__ beta,
                          float* __restrict__ out) {
    __shared__ float rs[8], rq[8];
    const int row = blockIdx.x;
    const int tid = threadIdx.x;
    const float* tr = t + (size_t)row * EMB;
    const float a = tr[tid], b = tr[tid + 256];
    float s = a + b, q = a * a + b * b;
#pragma unroll
    for (int off = 16; off > 0; off >>= 1) {
        s += __shfl_xor(s, off, 32);
        q += __shfl_xor(q, off, 32);
    }
    if ((tid & 31) == 0) { rs[tid >> 5] = s; rq[tid >> 5] = q; }
    __syncthreads();
    float st = 0.f, qt = 0.f;
#pragma unroll
    for (int i = 0; i < 8; ++i) { st += rs[i]; qt += rq[i]; }
    const float mu   = st * (1.0f / EMB);
    const float var  = qt * (1.0f / EMB) - mu * mu;
    const float rinv = rsqrtf(var + 1e-5f);
    out[(size_t)row * EMB + tid]       = (a - mu) * rinv * gamma[tid] + beta[tid];
    out[(size_t)row * EMB + tid + 256] = (b - mu) * rinv * gamma[tid + 256] + beta[tid + 256];
}

extern "C" void kernel_launch(void* const* d_in, const int* in_sizes, int n_in,
                              void* d_out, int out_size, void* d_ws, size_t ws_size,
                              hipStream_t stream) {
    const float* x     = (const float*)d_in[0];
    const float* Wq    = (const float*)d_in[1];
    const float* bq    = (const float*)d_in[2];
    const float* Wk    = (const float*)d_in[3];
    const float* bk    = (const float*)d_in[4];
    const float* Wv    = (const float*)d_in[5];
    const float* bv    = (const float*)d_in[6];
    const float* Wo    = (const float*)d_in[7];
    const float* bo    = (const float*)d_in[8];
    const float* gamma = (const float*)d_in[9];
    const float* beta  = (const float*)d_in[10];
    float* out = (float*)d_out;

    char* ws = (char*)d_ws;
    size_t off = 0;
    auto alloc = [&](size_t bytes) -> char* {
        char* p = ws + off;
        off += (bytes + 255) & ~(size_t)255;
        return p;
    };
    unsigned short* xb  = (unsigned short*)alloc((size_t)ROWS * EMB * 2);   // 8 MB
    unsigned short* WqT = (unsigned short*)alloc((size_t)EMB * EMB * 2);
    unsigned short* WkT = (unsigned short*)alloc((size_t)EMB * EMB * 2);
    unsigned short* WvT = (unsigned short*)alloc((size_t)EMB * EMB * 2);
    unsigned short* WoT = (unsigned short*)alloc((size_t)EMB * EMB * 2);
    unsigned short* Qh  = (unsigned short*)alloc((size_t)ROWS * EMB * 2);   // 8 MB
    unsigned short* Kh  = (unsigned short*)alloc((size_t)ROWS * EMB * 2);   // 8 MB
    unsigned short* Vt  = (unsigned short*)alloc((size_t)ROWS * EMB * 2);   // 8 MB
    unsigned short* ctx = (unsigned short*)alloc((size_t)ROWS * EMB * 2);   // 8 MB
    float* tbuf = (float*)Qh;  // 16 MB f32 aliased over Q+K (dead after attention)

    cvt_bf16_kernel<<<4096, 256, 0, stream>>>(x, xb, ROWS * EMB);
    cvtT512_kernel<<<EMB, 256, 0, stream>>>(Wq, WqT);
    cvtT512_kernel<<<EMB, 256, 0, stream>>>(Wk, WkT);
    cvtT512_kernel<<<EMB, 256, 0, stream>>>(Wv, WvT);
    cvtT512_kernel<<<EMB, 256, 0, stream>>>(Wo, WoT);

    const dim3 gg(512), gb(128);   // 2048 waves: 256 m-tiles x 8 n-groups
    gemm512_kernel<<<gg, gb, 0, stream>>>(xb, WqT, bq, nullptr, Qh, nullptr, 0);
    gemm512_kernel<<<gg, gb, 0, stream>>>(xb, WkT, bk, nullptr, Kh, nullptr, 0);
    gemm512_kernel<<<gg, gb, 0, stream>>>(xb, WvT, bv, nullptr, Vt, nullptr, 1);
    attn_kernel<<<2048, 64, 0, stream>>>(Qh, Kh, Vt, ctx);
    gemm512_kernel<<<gg, gb, 0, stream>>>(ctx, WoT, bo, x, nullptr, tbuf, 2);
    ln_kernel<<<ROWS, 256, 0, stream>>>(tbuf, gamma, beta, out);
}
